// GatedSpectralConv_49589692400197
// MI455X (gfx1250) — matrix operand order repaired and checked by reading, not compiled, not run on hardware
//
#include <hip/hip_runtime.h>
#include <math.h>

// ---------------------------------------------------------------------------
// GatedSpectralConv for MI455X (gfx1250, wave32, WMMA)
//   B=4, L=4096, D=1024, F=512
//   GEMM1: Z[16384,1536] = x[16384,1024] @ [W_spec | W_decay]  (+bias on decay)
//   scan : h_t = sigmoid(logit)*rot * h_{t-1} + u_t   (2048 lanes, L steps)
//   GEMM2: out[16384,1024] = y[16384,1024] @ W_out
// GEMMs use v_wmma_f32_16x16x32_bf16 (bf16 in, f32 accumulate).
// Staging uses GLOBAL_LOAD_ASYNC_TO_LDS_B128 when the builtin exists.
// ---------------------------------------------------------------------------

typedef __bf16 bf16_t;
typedef __attribute__((ext_vector_type(16))) __bf16 v16bf;
typedef __attribute__((ext_vector_type(8)))  float  v8f;
typedef __attribute__((ext_vector_type(4)))  float  f32x4;
typedef __attribute__((ext_vector_type(4)))  int    v4i;

#define CB   4
#define CL   4096
#define CD   1024
#define CF   512
#define CM   (CB * CL)        // 16384 rows
#define CK   CD               // 1024 reduction
#define CN1  (3 * CF)         // 1536: [u_re | u_im | decay_logit]
#define CN2  CD               // 1024

// LDS tiles: rows x 32 k, padded to 40 elems/row (80B) to spread banks.
#define LDP 40

#if __has_builtin(__builtin_amdgcn_global_load_async_to_lds_b128)
#define USE_ASYNC_LDS 1
#endif

#ifdef USE_ASYNC_LDS
typedef __attribute__((address_space(1))) v4i gas_v4i;   // global
typedef __attribute__((address_space(3))) v4i las_v4i;   // LDS

__device__ __forceinline__ void async_cp16(const bf16_t* g, bf16_t* l)
{
    // global (AS1) -> LDS (AS3), 16 bytes per lane, tracked by ASYNCcnt
    __builtin_amdgcn_global_load_async_to_lds_b128(
        (gas_v4i*)const_cast<bf16_t*>(g),
        (las_v4i*)l,
        0, 0);
}
__device__ __forceinline__ void wait_async0()
{
#if __has_builtin(__builtin_amdgcn_s_wait_asynccnt)
    __builtin_amdgcn_s_wait_asynccnt(0);
#else
    asm volatile("s_wait_asynccnt 0x0" ::: "memory");
#endif
}
#endif

// ---------------------------------------------------------------------------
// Tiled bf16 WMMA GEMM:  C[M,N] = A[M,K] * Bt[N,K]^T  (+ optional bias[n])
// Block = 256 threads (8 waves), block tile 256x128, wave tile 64x64
// (4x4 = 16 v_wmma_f32_16x16x32_bf16 per wave per K-step of 32).
// ---------------------------------------------------------------------------
__global__ __launch_bounds__(256)
void gemm_bf16_wmma(const bf16_t* __restrict__ A,
                    const bf16_t* __restrict__ Bt,
                    const float*  __restrict__ bias,
                    float*        __restrict__ C,
                    int M, int N, int K)
{
    __shared__ bf16_t lsA[256 * LDP];   // 20.0 KB
    __shared__ bf16_t lsB[128 * LDP];   // 10.0 KB

    const int tid   = threadIdx.x;
    const int lane  = tid & 31;
    const int wid   = tid >> 5;
    const int waveM = wid & 3;          // 0..3 -> 64-row slab
    const int waveN = wid >> 2;         // 0..1 -> 64-col slab
    const int m0    = blockIdx.x * 256;
    const int n0    = blockIdx.y * 128;

    const int mlane = lane & 15;        // row/col within 16x16 tile
    const int hlane = lane >> 4;        // half-wave selector

    v8f acc[4][4] = {};                 // 4 (M) x 4 (N) 16x16 f32 accumulators

    for (int k0 = 0; k0 < K; k0 += 32) {
        // ---- stage tiles into LDS: A 256x32 (1024 chunks), B 128x32 (512) --
#ifdef USE_ASYNC_LDS
        #pragma unroll
        for (int i = 0; i < 4; ++i) {
            const int c = tid + 256 * i, row = c >> 2, q = c & 3;
            async_cp16(A + (size_t)(m0 + row) * K + k0 + q * 8,
                       &lsA[row * LDP + q * 8]);
        }
        #pragma unroll
        for (int i = 0; i < 2; ++i) {
            const int c = tid + 256 * i, row = c >> 2, q = c & 3;
            async_cp16(Bt + (size_t)(n0 + row) * K + k0 + q * 8,
                       &lsB[row * LDP + q * 8]);
        }
        wait_async0();
#else
        #pragma unroll
        for (int i = 0; i < 4; ++i) {
            const int c = tid + 256 * i, row = c >> 2, q = c & 3;
            *(f32x4*)&lsA[row * LDP + q * 8] =
                *(const f32x4*)(A + (size_t)(m0 + row) * K + k0 + q * 8);
        }
        #pragma unroll
        for (int i = 0; i < 2; ++i) {
            const int c = tid + 256 * i, row = c >> 2, q = c & 3;
            *(f32x4*)&lsB[row * LDP + q * 8] =
                *(const f32x4*)(Bt + (size_t)(n0 + row) * K + k0 + q * 8);
        }
#endif
        __syncthreads();

        // ---- build WMMA fragments per the CDNA5 16-bit layouts ----
        union FragU { v16bf v; f32x4 f[2]; };
        FragU af[4], bfg[4];

        #pragma unroll
        for (int sm = 0; sm < 4; ++sm) {
            const int mrow = waveM * 64 + sm * 16 + mlane;
            // A 16x32: elems 0..7 -> K = h*8..h*8+7 ; 8..15 -> K = 16+h*8..+7
            af[sm].f[0] = *(const f32x4*)&lsA[mrow * LDP + hlane * 8];
            af[sm].f[1] = *(const f32x4*)&lsA[mrow * LDP + 16 + hlane * 8];
        }
        #pragma unroll
        for (int sn = 0; sn < 4; ++sn) {
            const int nrow = waveN * 64 + sn * 16 + mlane;
            // B 32x16 (stored transposed, N-major): elems 0..7 -> K = h*8..h*8+7 ; 8..15 -> K = 16+h*8..+7
            bfg[sn].f[0] = *(const f32x4*)&lsB[nrow * LDP + hlane * 8];
            bfg[sn].f[1] = *(const f32x4*)&lsB[nrow * LDP + 16 + hlane * 8];
        }

        // ---- 16 WMMAs per wave per K-step ----
        #pragma unroll
        for (int sm = 0; sm < 4; ++sm)
            #pragma unroll
            for (int sn = 0; sn < 4; ++sn)
                acc[sm][sn] = __builtin_amdgcn_wmma_f32_16x16x32_bf16(
                    false, af[sm].v, false, bfg[sn].v,
                    (short)0, acc[sm][sn], false, false);

        __syncthreads();
    }

    // ---- epilogue: C layout VGPR r -> M = r + 8*h, N = lane&15 ----
    #pragma unroll
    for (int sm = 0; sm < 4; ++sm) {
        #pragma unroll
        for (int sn = 0; sn < 4; ++sn) {
            const int gn = n0 + waveN * 64 + sn * 16 + mlane;
            const float bv = bias ? bias[gn] : 0.0f;
            #pragma unroll
            for (int r = 0; r < 8; ++r) {
                const int gm = m0 + waveM * 64 + sm * 16 + hlane * 8 + r;
                C[(size_t)gm * N + gn] = acc[sm][sn][r] + bv;
            }
        }
    }
}

// ---------------------------------------------------------------------------
// Sequential recurrence: one thread per (b,f); software-pipelined loads.
// Z row layout (per time step): [u_re(512) | u_im(512) | decay_logit(512)]
// Writes Y (bf16, [Re|Im] per row) for GEMM2 and final h to hfin.
// ---------------------------------------------------------------------------
__global__ __launch_bounds__(256)
void scan_kernel(const float* __restrict__ Z,
                 const float* __restrict__ freqs,
                 bf16_t*      __restrict__ Y,
                 float*       __restrict__ hfin)
{
    const int tid = blockIdx.x * blockDim.x + threadIdx.x;   // 0..B*F-1
    const int b = tid / CF;
    const int f = tid % CF;

    const float w  = freqs[f] * 0.1f;
    const float cw = cosf(w);
    const float sw = sinf(w);

    const float* zb = Z + (size_t)b * CL * CN1;

    float hr = 0.0f, hi = 0.0f;
    float ur = zb[f], ui = zb[CF + f], lg = zb[2 * CF + f];

    for (int t = 0; t < CL; ++t) {
        // prefetch next step's operands while computing this one
        float nur = 0.0f, nui = 0.0f, nlg = 0.0f;
        if (t + 1 < CL) {
            const float* zn = zb + (size_t)(t + 1) * CN1;
            nur = zn[f]; nui = zn[CF + f]; nlg = zn[2 * CF + f];
        }

        const float d  = 1.0f / (1.0f + expf(-lg));
        const float ar = d * cw;
        const float ai = d * sw;
        const float tr = fmaf(ar, hr, fmaf(-ai, hi, ur));
        const float ti = fmaf(ar, hi, fmaf( ai, hr, ui));
        hr = tr; hi = ti;

        bf16_t* yrow = Y + (size_t)(b * CL + t) * (2 * CF);
        yrow[f]      = (bf16_t)hr;
        yrow[CF + f] = (bf16_t)hi;

        ur = nur; ui = nui; lg = nlg;
    }

    hfin[b * CF + f]           = hr;   // h_final.real
    hfin[CB * CF + b * CF + f] = hi;   // h_final.imag
}

// ---------------------------------------------------------------------------
// Prep kernels: fp32 -> bf16, weight transposes (N-major), bias pack.
// ---------------------------------------------------------------------------
__global__ void cvt_x_kernel(const float* __restrict__ x,
                             bf16_t* __restrict__ xb, int n)
{
    for (int i = blockIdx.x * blockDim.x + threadIdx.x; i < n;
         i += gridDim.x * blockDim.x)
        xb[i] = (bf16_t)x[i];
}

__global__ void pack_w1_kernel(const float* __restrict__ Wspec,
                               const float* __restrict__ Wdec,
                               const float* __restrict__ bdec,
                               bf16_t* __restrict__ w1t,
                               float*  __restrict__ bias)
{
    const int n = CN1 * CK;
    for (int i = blockIdx.x * blockDim.x + threadIdx.x; i < n;
         i += gridDim.x * blockDim.x) {
        const int col = i / CK;        // output column n (0..1535)
        const int k   = i % CK;
        const float v = (col < 2 * CF)
            ? Wspec[(size_t)k * (2 * CF) + col]
            : Wdec [(size_t)k * CF + (col - 2 * CF)];
        w1t[i] = (bf16_t)v;            // row-major [N][K]
    }
    const int i0 = blockIdx.x * blockDim.x + threadIdx.x;
    if (i0 < CN1) bias[i0] = (i0 < 2 * CF) ? 0.0f : bdec[i0 - 2 * CF];
}

__global__ void pack_wout_kernel(const float* __restrict__ Wout,
                                 bf16_t* __restrict__ wot)
{
    const int n = CN2 * (2 * CF);
    for (int i = blockIdx.x * blockDim.x + threadIdx.x; i < n;
         i += gridDim.x * blockDim.x) {
        const int d = i / (2 * CF);    // output column d
        const int f = i % (2 * CF);
        wot[i] = (bf16_t)Wout[(size_t)f * CD + d];   // row-major [N=D][K=2F]
    }
}

// ---------------------------------------------------------------------------
extern "C" void kernel_launch(void* const* d_in, const int* in_sizes, int n_in,
                              void* d_out, int out_size, void* d_ws, size_t ws_size,
                              hipStream_t stream)
{
    (void)in_sizes; (void)n_in; (void)out_size; (void)ws_size;

    const float* x     = (const float*)d_in[0];
    const float* Wspec = (const float*)d_in[1];
    const float* Wdec  = (const float*)d_in[2];
    const float* bdec  = (const float*)d_in[3];
    const float* freqs = (const float*)d_in[4];
    const float* Wout  = (const float*)d_in[5];
    float* out = (float*)d_out;

    char* ws = (char*)d_ws;
    size_t off = 0;
    bf16_t* xb   = (bf16_t*)(ws + off); off += (size_t)CM  * CK  * 2;  // 32 MB
    bf16_t* w1t  = (bf16_t*)(ws + off); off += (size_t)CN1 * CK  * 2;  //  3 MB
    bf16_t* wot  = (bf16_t*)(ws + off); off += (size_t)CN2 * CK  * 2;  //  2 MB
    float*  bias = (float*) (ws + off); off += (size_t)2048 * 4;
    float*  Z    = (float*) (ws + off); off += (size_t)CM  * CN1 * 4;  // 96 MB
    bf16_t* Y    = (bf16_t*)(ws + off); off += (size_t)CM  * CK  * 2;  // 32 MB

    // prep: convert/transposes
    cvt_x_kernel  <<<2048, 256, 0, stream>>>(x, xb, CM * CK);
    pack_w1_kernel<<<1024, 256, 0, stream>>>(Wspec, Wdec, bdec, w1t, bias);
    pack_wout_kernel<<<1024, 256, 0, stream>>>(Wout, wot);

    // GEMM1: Z = x @ [W_spec | W_decay] (+bias)
    dim3 g1(CM / 256, CN1 / 128);
    gemm_bf16_wmma<<<g1, 256, 0, stream>>>(xb, w1t, bias, Z, CM, CN1, CK);

    // recurrence over time; writes Y (bf16) and h_final tail of d_out
    scan_kernel<<<(CB * CF) / 256, 256, 0, stream>>>(
        Z, freqs, Y, out + (size_t)CM * CD);

    // GEMM2: out = y @ W_out
    dim3 g2(CM / 256, CN2 / 128);
    gemm_bf16_wmma<<<g2, 256, 0, stream>>>(Y, wot, nullptr, out, CM, CN2, CK);
}
